// GAT_72000831750144
// MI455X (gfx1250) — matrix operand order repaired and checked by reading, not compiled, not run on hardware
//
#include <hip/hip_runtime.h>
#include <math.h>

#define NNODES 50000
#define NEDGES 800000
#define NEGSLOPE 0.2f
#define ENC_NEG_INF 0x007FFFFFu

typedef __attribute__((ext_vector_type(16))) _Float16 v16h;
typedef __attribute__((ext_vector_type(8)))  float    v8f;

// order-preserving float <-> uint map so atomicMax(uint) == float max
__device__ __forceinline__ unsigned enc_f32(float f) {
    unsigned u = __float_as_uint(f);
    return u ^ (unsigned)(((int)u >> 31) | 0x80000000);
}
__device__ __forceinline__ float dec_f32(unsigned u) {
    unsigned mask = (u & 0x80000000u) ? 0x80000000u : 0xFFFFFFFFu;
    return __uint_as_float(u ^ mask);
}
__device__ __forceinline__ float lrelu(float v) { return v > 0.0f ? v : NEGSLOPE * v; }

__global__ void fill_u32(unsigned* __restrict__ p, int n, unsigned v) {
    int i = blockIdx.x * blockDim.x + threadIdx.x;
    if (i < n) p[i] = v;
}

// C[M x KOUT] = A[M x KIN] @ W[KIN x KOUT], row-major fp32 in/out.
// One block per 16-row slab; wave w computes column tile w.  Both operands are
// staged once per block into LDS, converted to f16 and pre-swizzled into the
// CDNA5 WMMA fragment layout (ISA 7.12.2, wave32), so the inner loop is pure
// ds_load_b128 -> v_wmma_f32_16x16x32_f16 (fp32 accumulate).
template <int KIN, int KOUT>
__global__ void gemm_wmma_f16(const float* __restrict__ A, const float* __restrict__ W,
                              float* __restrict__ C, int M) {
    constexpr int KS = KIN / 32;   // WMMA K-steps
    constexpr int CT = KOUT / 16;  // column tiles == waves per block
    __shared__ __align__(32) _Float16 ldsA[KS * 32 * 16];        // A fragments (shared by all waves)
    __shared__ __align__(32) _Float16 ldsB[KS * CT * 32 * 16];   // B fragments per column tile

    const int tid = threadIdx.x;
    const int rt  = blockIdx.x;          // 16-row tile index
    if (rt * 16 >= M) return;

    // stage A tile (16 x KIN): lane<16 holds row=lane, K in {0..7,16..23} of its 32-chunk;
    // lane>=16 holds K in {8..15,24..31}
    for (int t = tid; t < KS * 32 * 16; t += blockDim.x) {
        const int i   = t & 15;
        const int ln  = (t >> 4) & 31;
        const int ks  = t >> 9;
        const int k   = ks * 32 + ((ln >= 16) ? 8 : 0) + (i < 8 ? i : i + 8);
        const int row = rt * 16 + (ln & 15);
        ldsA[t] = (_Float16)A[(size_t)row * KIN + k];
    }
    // stage W (KIN x KOUT): lane<16 holds K in {0..7,16..23} of col n, lane>=16 holds K in {8..15,24..31}
    for (int t = tid; t < KS * CT * 32 * 16; t += blockDim.x) {
        const int i  = t & 15;
        const int ln = (t >> 4) & 31;
        const int ct = (t >> 9) % CT;
        const int ks = t / (512 * CT);
        const int k  = ks * 32 + ((ln >= 16) ? 8 : 0) + (i < 8 ? i : i + 8);
        const int n  = ct * 16 + (ln & 15);
        ldsB[t] = (_Float16)W[(size_t)k * KOUT + n];
    }
    __syncthreads();

    const int wv   = tid >> 5;   // this wave's column tile
    const int lane = tid & 31;
    v8f acc = {0.f, 0.f, 0.f, 0.f, 0.f, 0.f, 0.f, 0.f};
#pragma unroll
    for (int ks = 0; ks < KS; ++ks) {
        v16h a = *(const v16h*)&ldsA[(ks * 32 + lane) * 16];
        v16h b = *(const v16h*)&ldsB[((ks * CT + wv) * 32 + lane) * 16];
        acc = __builtin_amdgcn_wmma_f32_16x16x32_f16(false, a, false, b, (short)0, acc,
                                                     false, false);
    }
    // D 16x16 f32: VGPR r -> row (hi?8:0)+r, col n
    const int n  = wv * 16 + (lane & 15);
    const int mb = rt * 16 + ((lane >= 16) ? 8 : 0);
#pragma unroll
    for (int r = 0; r < 8; ++r) C[(size_t)(mb + r) * KOUT + n] = acc[r];
}

// al_src[n,h] = <h[n,h,:], a_src[h,:]>, al_dst likewise. One thread per (node, head).
__global__ void att_logits(const float* __restrict__ h, const float* __restrict__ asrc,
                           const float* __restrict__ adst, float* __restrict__ als,
                           float* __restrict__ ald, int N, int H, int C) {
    int t = blockIdx.x * blockDim.x + threadIdx.x;
    if (t >= N * H) return;
    int n = t / H, hh = t % H;
    const float* row = h + (size_t)n * (H * C) + (size_t)hh * C;
    const float* as  = asrc + hh * C;
    const float* ad  = adst + hh * C;
    float ss = 0.f, sd = 0.f;
    for (int c = 0; c < C; ++c) {
        float x = row[c];
        ss += x * as[c];
        sd += x * ad[c];
    }
    als[t] = ss;
    ald[t] = sd;
}

__device__ __forceinline__ void edge_sd(const long long* __restrict__ esrc,
                                        const long long* __restrict__ edst,
                                        int e, int E, int& s, int& d) {
    if (e < E) { s = (int)esrc[e]; d = (int)edst[e]; }
    else       { s = e - E; d = s; }   // appended self-loops
}

// pass 1: segment max of leaky_relu(al_s[src]+al_d[dst]) per (dst, head)
__global__ void edge_logit_max(const long long* __restrict__ esrc,
                               const long long* __restrict__ edst, int E, int N, int H,
                               const float* __restrict__ als, const float* __restrict__ ald,
                               unsigned* __restrict__ menc) {
    int e = blockIdx.x * blockDim.x + threadIdx.x;
    if (e >= E + N) return;
    int s, d;
    edge_sd(esrc, edst, e, E, s, d);
    for (int h = 0; h < H; ++h) {
        float v = lrelu(als[s * H + h] + ald[d * H + h]);
        atomicMax(&menc[d * H + h], enc_f32(v));
    }
}

// pass 2: denom[dst,h] += exp(e - max)
__global__ void edge_expsum(const long long* __restrict__ esrc,
                            const long long* __restrict__ edst, int E, int N, int H,
                            const float* __restrict__ als, const float* __restrict__ ald,
                            const unsigned* __restrict__ menc, float* __restrict__ den) {
    int e = blockIdx.x * blockDim.x + threadIdx.x;
    if (e >= E + N) return;
    int s, d;
    edge_sd(esrc, edst, e, E, s, d);
    for (int h = 0; h < H; ++h) {
        float v  = lrelu(als[s * H + h] + ald[d * H + h]);
        float ex = __expf(v - dec_f32(menc[d * H + h]));
        atomicAdd(&den[d * H + h], ex);
    }
}

// pass 3 layer-1: agg[dst, :] += alpha * h[src, :]  (H=4, C=32; wave per edge, float4/lane)
__global__ void edge_agg_l1(const long long* __restrict__ esrc,
                            const long long* __restrict__ edst, int E, int N,
                            const float* __restrict__ als, const float* __restrict__ ald,
                            const unsigned* __restrict__ menc, const float* __restrict__ den,
                            const float* __restrict__ hfeat, float* __restrict__ agg) {
    int gid = blockIdx.x * blockDim.x + threadIdx.x;
    int w = gid >> 5, lane = gid & 31;
    if (w >= E + N) return;
    int s, d;
    edge_sd(esrc, edst, w, E, s, d);
    int h = lane >> 3;                        // channels 4*lane..4*lane+3 -> head lane/8
    float v     = lrelu(als[s * 4 + h] + ald[d * 4 + h]);
    float alpha = __expf(v - dec_f32(menc[d * 4 + h])) / den[d * 4 + h];
    const float* hs = hfeat + (size_t)s * 128 + lane * 4;
    float*       ad = agg   + (size_t)d * 128 + lane * 4;
#pragma unroll
    for (int i = 0; i < 4; ++i) atomicAdd(&ad[i], hs[i] * alpha);
}

// pass 3 layer-2: H=1, C=64; wave per edge, 2 floats/lane
__global__ void edge_agg_l2(const long long* __restrict__ esrc,
                            const long long* __restrict__ edst, int E, int N,
                            const float* __restrict__ als, const float* __restrict__ ald,
                            const unsigned* __restrict__ menc, const float* __restrict__ den,
                            const float* __restrict__ hfeat, float* __restrict__ agg) {
    int gid = blockIdx.x * blockDim.x + threadIdx.x;
    int w = gid >> 5, lane = gid & 31;
    if (w >= E + N) return;
    int s, d;
    edge_sd(esrc, edst, w, E, s, d);
    float v     = lrelu(als[s] + ald[d]);
    float alpha = __expf(v - dec_f32(menc[d])) / den[d];
    const float* hs = hfeat + (size_t)s * 64 + lane * 2;
    float*       ad = agg   + (size_t)d * 64 + lane * 2;
    atomicAdd(&ad[0], hs[0] * alpha);
    atomicAdd(&ad[1], hs[1] * alpha);
}

// h = relu(agg + b), 128 channels
__global__ void relu_bias(float* __restrict__ h, const float* __restrict__ agg,
                          const float* __restrict__ b, int N) {
    int i = blockIdx.x * blockDim.x + threadIdx.x;
    if (i >= N * 128) return;
    float v = agg[i] + b[i & 127];
    h[i] = v > 0.f ? v : 0.f;
}

// out[n,:] = log_softmax(agg2[n,:] + b2) over 64 channels; wave32 per node
__global__ void finalize_lsm(const float* __restrict__ agg, const float* __restrict__ b,
                             float* __restrict__ out, int N) {
    int gid = blockIdx.x * blockDim.x + threadIdx.x;
    int n = gid >> 5, lane = gid & 31;
    if (n >= N) return;
    float v0 = agg[(size_t)n * 64 + lane]      + b[lane];
    float v1 = agg[(size_t)n * 64 + 32 + lane] + b[32 + lane];
    float mx = fmaxf(v0, v1);
#pragma unroll
    for (int o = 16; o > 0; o >>= 1) mx = fmaxf(mx, __shfl_xor(mx, o, 32));
    float se = __expf(v0 - mx) + __expf(v1 - mx);
#pragma unroll
    for (int o = 16; o > 0; o >>= 1) se += __shfl_xor(se, o, 32);
    float lse = mx + __logf(se);
    out[(size_t)n * 64 + lane]      = v0 - lse;
    out[(size_t)n * 64 + 32 + lane] = v1 - lse;
}

extern "C" void kernel_launch(void* const* d_in, const int* in_sizes, int n_in,
                              void* d_out, int out_size, void* d_ws, size_t ws_size,
                              hipStream_t stream) {
    const float*     x     = (const float*)d_in[0];
    const long long* ei    = (const long long*)d_in[1];   // int64 [2, E]
    const float*     W1    = (const float*)d_in[2];
    const float*     asrc1 = (const float*)d_in[3];
    const float*     adst1 = (const float*)d_in[4];
    const float*     b1    = (const float*)d_in[5];
    const float*     W2    = (const float*)d_in[6];
    const float*     asrc2 = (const float*)d_in[7];
    const float*     adst2 = (const float*)d_in[8];
    const float*     b2    = (const float*)d_in[9];

    const int N = NNODES, E = NEDGES;
    const long long* esrc = ei;
    const long long* edst = ei + E;

    // workspace layout (floats / u32), ~81 MB total
    float*    h1   = (float*)d_ws;                 // N*128
    float*    als1 = h1 + (size_t)N * 128;         // N*4
    float*    ald1 = als1 + (size_t)N * 4;         // N*4
    unsigned* m1   = (unsigned*)(ald1 + (size_t)N * 4);  // N*4
    float*    den1 = (float*)(m1 + (size_t)N * 4); // N*4
    float*    agg1 = den1 + (size_t)N * 4;         // N*128
    float*    h2   = agg1 + (size_t)N * 128;       // N*64
    float*    als2 = h2 + (size_t)N * 64;          // N
    float*    ald2 = als2 + N;                     // N
    unsigned* m2   = (unsigned*)(ald2 + N);        // N
    float*    den2 = (float*)(m2 + N);             // N
    float*    agg2 = den2 + N;                     // N*64
    float*    outp = (float*)d_out;

    auto cdiv = [](long long a, long long b) { return (int)((a + b - 1) / b); };
    const int tot = E + N;   // edges incl. self loops

    // re-init accumulators every call (graph-replay safe)
    fill_u32<<<cdiv(N * 4, 256), 256, 0, stream>>>(m1, N * 4, ENC_NEG_INF);
    fill_u32<<<cdiv(N * 4, 256), 256, 0, stream>>>((unsigned*)den1, N * 4, 0u);
    fill_u32<<<cdiv((long long)N * 128, 256), 256, 0, stream>>>((unsigned*)agg1, N * 128, 0u);
    fill_u32<<<cdiv(N, 256), 256, 0, stream>>>(m2, N, ENC_NEG_INF);
    fill_u32<<<cdiv(N, 256), 256, 0, stream>>>((unsigned*)den2, N, 0u);
    fill_u32<<<cdiv((long long)N * 64, 256), 256, 0, stream>>>((unsigned*)agg2, N * 64, 0u);

    // ---- layer 1 ----
    gemm_wmma_f16<128, 128><<<N / 16, 8 * 32, 0, stream>>>(x, W1, h1, N);
    att_logits<<<cdiv(N * 4, 256), 256, 0, stream>>>(h1, asrc1, adst1, als1, ald1, N, 4, 32);
    edge_logit_max<<<cdiv(tot, 256), 256, 0, stream>>>(esrc, edst, E, N, 4, als1, ald1, m1);
    edge_expsum<<<cdiv(tot, 256), 256, 0, stream>>>(esrc, edst, E, N, 4, als1, ald1, m1, den1);
    edge_agg_l1<<<cdiv((long long)tot * 32, 256), 256, 0, stream>>>(esrc, edst, E, N, als1,
                                                                    ald1, m1, den1, h1, agg1);
    relu_bias<<<cdiv((long long)N * 128, 256), 256, 0, stream>>>(h1, agg1, b1, N);

    // ---- layer 2 ----
    gemm_wmma_f16<128, 64><<<N / 16, 4 * 32, 0, stream>>>(h1, W2, h2, N);
    att_logits<<<cdiv(N, 256), 256, 0, stream>>>(h2, asrc2, adst2, als2, ald2, N, 1, 64);
    edge_logit_max<<<cdiv(tot, 256), 256, 0, stream>>>(esrc, edst, E, N, 1, als2, ald2, m2);
    edge_expsum<<<cdiv(tot, 256), 256, 0, stream>>>(esrc, edst, E, N, 1, als2, ald2, m2, den2);
    edge_agg_l2<<<cdiv((long long)tot * 32, 256), 256, 0, stream>>>(esrc, edst, E, N, als2,
                                                                    ald2, m2, den2, h2, agg2);
    finalize_lsm<<<cdiv((long long)N * 32, 256), 256, 0, stream>>>(agg2, b2, outp, N);
}
